// OuterProductMean_65635690217491
// MI455X (gfx1250) — matrix operand order repaired and checked by reading, not compiled, not run on hardware
//
#include <hip/hip_runtime.h>

typedef __attribute__((ext_vector_type(16))) _Float16 v16h;
typedef __attribute__((ext_vector_type(8)))  float    v8f;
typedef __attribute__((ext_vector_type(4)))  float    f32x4;

union Frag { f32x4 f[2]; v16h h; };

#define S_DIM 256
#define N_RES 256
#define C_M   256
#define C_H   32
#define C_Z   128

// ---------------------------------------------------------------------------
// Kernel 1: LayerNorm + dual 256->32 projections, masked, written transposed:
//   a_t[i][c][s], b_t[j][e][s]  (f16)  -- WMMA-fragment-friendly layout.
// One wave32 per (s, n) row. 8 waves / block.
// ---------------------------------------------------------------------------
__global__ __launch_bounds__(256) void opm_prep(
    const float* __restrict__ m, const float* __restrict__ mask,
    const float* __restrict__ ln_w, const float* __restrict__ ln_b,
    const float* __restrict__ w1, const float* __restrict__ b1,
    const float* __restrict__ w2, const float* __restrict__ b2,
    _Float16* __restrict__ a_t, _Float16* __restrict__ b_t)
{
    __shared__ float lnbuf[8][C_M];
    const int tid  = threadIdx.x;
    const int wave = tid >> 5;
    const int lane = tid & 31;
    const int row  = blockIdx.x * 8 + wave;       // row = s*256 + n
    const int s = row >> 8, n = row & 255;
    const float* x = m + (size_t)row * C_M;

    float xs[8], sum = 0.f, sq = 0.f;
    #pragma unroll
    for (int t = 0; t < 8; ++t) {
        float v = x[lane + t * 32];
        xs[t] = v; sum += v; sq += v * v;
    }
    #pragma unroll
    for (int off = 16; off >= 1; off >>= 1) {
        sum += __shfl_xor(sum, off, 32);
        sq  += __shfl_xor(sq,  off, 32);
    }
    const float mu  = sum * (1.f / C_M);
    const float var = sq * (1.f / C_M) - mu * mu;
    const float rs  = rsqrtf(var + 1e-5f);
    #pragma unroll
    for (int t = 0; t < 8; ++t) {
        int k = lane + t * 32;
        lnbuf[wave][k] = (xs[t] - mu) * rs * ln_w[k] + ln_b[k];
    }
    __syncthreads();

    const int h = lane;                            // c_h = 32 = wave width
    float acc1 = b1[h], acc2 = b2[h];
    for (int k = 0; k < C_M; ++k) {
        float lv = lnbuf[wave][k];
        acc1 += lv * w1[k * C_H + h];
        acc2 += lv * w2[k * C_H + h];
    }
    const float mk = mask[s * N_RES + n];
    const size_t oi = ((size_t)n * C_H + h) * S_DIM + s;
    a_t[oi] = (_Float16)(acc1 * mk);
    b_t[oi] = (_Float16)(acc2 * mk);
}

// ---------------------------------------------------------------------------
// Kernel 2: norm[i][j] = sum_s mask[s,i]*mask[s,j] + 1e-3
// ---------------------------------------------------------------------------
__global__ __launch_bounds__(256) void opm_norm(
    const float* __restrict__ mask, float* __restrict__ nrm)
{
    const int i = blockIdx.x, j = threadIdx.x;
    float acc = 0.f;
    for (int t = 0; t < S_DIM; ++t)
        acc += mask[t * N_RES + i] * mask[t * N_RES + j];
    nrm[i * N_RES + j] = acc + 1e-3f;
}

// ---------------------------------------------------------------------------
// Kernel 3: Wt[z][k] = (f16) w_out[k][z]   (transpose + convert, 128x1024)
// ---------------------------------------------------------------------------
__global__ __launch_bounds__(256) void opm_wt(
    const float* __restrict__ w_out, _Float16* __restrict__ Wt)
{
    const int idx = blockIdx.x * 256 + threadIdx.x;   // 131072 total
    const int z = idx >> 10, k = idx & 1023;
    Wt[idx] = (_Float16)w_out[k * C_Z + z];
}

// ---------------------------------------------------------------------------
// Kernel 4: fused outer-product + projection. One block = 4x4 (i,j) pairs.
// Stage A: per pair, 32x32 outer block = 2x2 WMMA tiles over K=S=256.
// Stage B: 16 pairs x 1024 (f16 in LDS) x Wt -> 16 x 128, bias + /norm.
// ---------------------------------------------------------------------------
__global__ __launch_bounds__(256) void opm_main(
    const _Float16* __restrict__ a_t, const _Float16* __restrict__ b_t,
    const _Float16* __restrict__ Wt, const float* __restrict__ nrm,
    const float* __restrict__ b_out, float* __restrict__ out)
{
    __shared__ _Float16 Os[16 * 1024];     // 32 KB: 16 pairs x vec(32x32)
    const int tid  = threadIdx.x;
    const int wave = tid >> 5;
    const int lane = tid & 31;
    const int half = lane >> 4;
    const int ln16 = lane & 15;
    const int i0 = (blockIdx.x >> 6) * 4;
    const int j0 = (blockIdx.x & 63) * 4;

    // -------------------- Stage A: outer blocks -> Os --------------------
    #pragma unroll
    for (int pp = 0; pp < 2; ++pp) {
        const int p  = wave * 2 + pp;              // pair 0..15
        const int gi = i0 + (p >> 2);
        const int gj = j0 + (p & 3);
        const _Float16* Abase = a_t + (size_t)gi * C_H * S_DIM;
        const _Float16* Bbase = b_t + (size_t)gj * C_H * S_DIM;

        v8f acc[4] = {};                           // [ct*2+et]
        #pragma unroll
        for (int ks = 0; ks < 8; ++ks) {           // K = 256 in steps of 32
            Frag fa[2], fb[2];
            #pragma unroll
            for (int ct = 0; ct < 2; ++ct) {
                // A (MxK) frag: elem t<8 -> K=half*8+t ; t>=8 -> K=16+half*8+t-8
                const _Float16* pa =
                    Abase + (size_t)(ct * 16 + ln16) * S_DIM + ks * 32 + half * 8;
                fa[ct].f[0] = *(const f32x4*)pa;
                fa[ct].f[1] = *(const f32x4*)(pa + 16);
            }
            #pragma unroll
            for (int et = 0; et < 2; ++et) {
                // B (KxN) frag: elem t<8 -> K=half*8+t ; t>=8 -> K=16+half*8+t-8, N=ln16
                const _Float16* pb =
                    Bbase + (size_t)(et * 16 + ln16) * S_DIM + ks * 32 + half * 8;
                fb[et].f[0] = *(const f32x4*)pb;
                fb[et].f[1] = *(const f32x4*)(pb + 16);
            }
            #pragma unroll
            for (int ct = 0; ct < 2; ++ct)
                #pragma unroll
                for (int et = 0; et < 2; ++et)
                    acc[ct * 2 + et] = __builtin_amdgcn_wmma_f32_16x16x32_f16(
                        false, fa[ct].h, false, fb[et].h,
                        (short)0, acc[ct * 2 + et], false, false);
        }
        // spill outer block to LDS as f16: k = c*32 + e
        #pragma unroll
        for (int ct = 0; ct < 2; ++ct)
            #pragma unroll
            for (int et = 0; et < 2; ++et)
                #pragma unroll
                for (int r = 0; r < 8; ++r) {
                    const int c = ct * 16 + half * 8 + r;   // C/D: M = r + half*8
                    const int e = et * 16 + ln16;
                    Os[p * 1024 + c * 32 + e] = (_Float16)acc[ct * 2 + et][r];
                }
    }
    __syncthreads();

    // ------------- Stage B: [16 pairs] x 1024 @ Wt -> 16 x 128 -------------
    const int zt = wave;                           // one 16-wide z tile / wave
    v8f cz = {};
    #pragma unroll 4
    for (int kk = 0; kk < 32; ++kk) {              // K = 1024 in steps of 32
        Frag fa, fb;
        const _Float16* po = &Os[ln16 * 1024 + kk * 32 + half * 8];
        fa.f[0] = *(const f32x4*)po;
        fa.f[1] = *(const f32x4*)(po + 16);
        const _Float16* pw =
            Wt + (size_t)(zt * 16 + ln16) * 1024 + kk * 32 + half * 8;
        fb.f[0] = *(const f32x4*)pw;
        fb.f[1] = *(const f32x4*)(pw + 16);
        cz = __builtin_amdgcn_wmma_f32_16x16x32_f16(
            false, fa.h, false, fb.h, (short)0, cz, false, false);
    }

    const float bz = b_out[zt * 16 + ln16];
    #pragma unroll
    for (int r = 0; r < 8; ++r) {
        const int p  = half * 8 + r;               // C/D row = pair index
        const int gi = i0 + (p >> 2);
        const int gj = j0 + (p & 3);
        const float nv = nrm[gi * N_RES + gj];
        out[((size_t)(gi * N_RES + gj)) * C_Z + zt * 16 + ln16] =
            (cz[r] + bz) / nv;
    }
}

// ---------------------------------------------------------------------------
extern "C" void kernel_launch(void* const* d_in, const int* in_sizes, int n_in,
                              void* d_out, int out_size, void* d_ws, size_t ws_size,
                              hipStream_t stream) {
    const float* m     = (const float*)d_in[0];
    const float* mask  = (const float*)d_in[1];
    const float* ln_w  = (const float*)d_in[2];
    const float* ln_b  = (const float*)d_in[3];
    const float* w1    = (const float*)d_in[4];
    const float* b1    = (const float*)d_in[5];
    const float* w2    = (const float*)d_in[6];
    const float* b2    = (const float*)d_in[7];
    const float* w_out = (const float*)d_in[8];
    const float* b_out = (const float*)d_in[9];
    float* out = (float*)d_out;

    char* ws = (char*)d_ws;
    const size_t szAT = (size_t)N_RES * C_H * S_DIM * sizeof(_Float16); // 4 MB
    const size_t szWT = (size_t)C_Z * 1024 * sizeof(_Float16);          // 256 KB
    _Float16* a_t = (_Float16*)(ws);
    _Float16* b_t = (_Float16*)(ws + szAT);
    _Float16* Wt  = (_Float16*)(ws + 2 * szAT);
    float*    nrm = (float*)   (ws + 2 * szAT + szWT);

    opm_prep<<<8192, 256, 0, stream>>>(m, mask, ln_w, ln_b, w1, b1, w2, b2, a_t, b_t);
    opm_norm<<<256, 256, 0, stream>>>(mask, nrm);
    opm_wt<<<512, 256, 0, stream>>>(w_out, Wt);
    opm_main<<<4096, 256, 0, stream>>>(a_t, b_t, Wt, nrm, b_out, out);
}
